// GAT_54142357733553
// MI455X (gfx1250) — matrix operand order repaired and checked by reading, not compiled, not run on hardware
//
#include <hip/hip_runtime.h>
#include <math.h>

typedef __attribute__((ext_vector_type(16))) _Float16 v16h;
typedef __attribute__((ext_vector_type(8)))  float    v8f;

#define NEG_SLOPE 0.2f

// ---------- order-preserving float<->uint encoding for atomic max ----------
__device__ __forceinline__ unsigned enc_f32(float f) {
  unsigned u = __float_as_uint(f);
  return (u & 0x80000000u) ? ~u : (u | 0x80000000u);
}
__device__ __forceinline__ float dec_f32(unsigned u) {
  unsigned v = (u & 0x80000000u) ? (u & 0x7FFFFFFFu) : ~u;
  return __uint_as_float(v);
}
// enc(-inf) = ~0xFF800000
#define ENC_NEG_INF 0x007FFFFFu

// ---------- small utility kernels ----------
__global__ void fill_u32_kernel(unsigned* __restrict__ p, unsigned val, int n) {
  int i = blockIdx.x * blockDim.x + threadIdx.x;
  if (i < n) p[i] = val;
}

// X [M,K] fp32 row-major -> X16 [M,Kpad] f16 (zero padded)
__global__ void cvt_x_f16_kernel(const float* __restrict__ X, _Float16* __restrict__ X16,
                                 int M, int K, int Kpad) {
  int i = blockIdx.x * blockDim.x + threadIdx.x;
  if (i >= M * Kpad) return;
  int r = i / Kpad, k = i - r * Kpad;
  X16[i] = (k < K) ? (_Float16)X[(size_t)r * K + k] : (_Float16)0.0f;
}

// W [K,Ncols] fp32 row-major -> Wt [Ncols,Kpad] f16 (transposed, zero padded)
__global__ void cvt_wt_f16_kernel(const float* __restrict__ W, _Float16* __restrict__ Wt,
                                  int K, int Kpad, int Ncols) {
  int i = blockIdx.x * blockDim.x + threadIdx.x;
  if (i >= Ncols * Kpad) return;
  int c = i / Kpad, k = i - c * Kpad;
  Wt[i] = (k < K) ? (_Float16)W[(size_t)k * Ncols + c] : (_Float16)0.0f;
}

// ---------- WMMA GEMM: Out[M,Ncols] = X16[M,Kpad] * Wt16[Ncols,Kpad]^T ----------
// one wave per 16x16 output tile, K stepped by 32 through v_wmma_f32_16x16x32_f16
__global__ void __launch_bounds__(256)
gemm_wmma_f16_kernel(const _Float16* __restrict__ X, const _Float16* __restrict__ Wt,
                     float* __restrict__ Out, int M, int Kpad, int Ncols) {
  const int lane = threadIdx.x & 31;
  const int wave = threadIdx.x >> 5;
  const int ntiles = Ncols >> 4;
  const int mtiles = (M + 15) >> 4;
  const int tile = blockIdx.x * 8 + wave;
  if (tile >= mtiles * ntiles) return;           // wave-uniform
  const int tm = tile / ntiles;
  const int tn = tile - tm * ntiles;
  const int g  = lane >> 4;                      // lane group
  const int l15 = lane & 15;

  const int mr  = (tm << 4) + l15;
  const int row = (mr < M) ? mr : (M - 1);       // clamp loads
  const int col = (tn << 4) + l15;

  const _Float16* __restrict__ xrow = X  + (size_t)row * Kpad;
  const _Float16* __restrict__ wcol = Wt + (size_t)col * Kpad + g * 16;

  v8f acc = {};
  for (int kb = 0; kb < Kpad; kb += 32) {
    // A fragment: group g holds K = kb+16g+{0..15}, contiguous (matches B)
    const _Float16* xp = xrow + kb + g * 16;
    v16h a;
#pragma unroll
    for (int j = 0; j < 4; ++j) { a[2*j]   = xp[2*j];          a[2*j+1]   = xp[2*j+1]; }
#pragma unroll
    for (int j = 0; j < 4; ++j) { a[8+2*j] = xp[8 + 2*j];      a[8+2*j+1] = xp[8 + 2*j + 1]; }
    // B fragment: column n=lane&15, group g holds K = kb+16g+{0..15}, contiguous in Wt
    const _Float16* wp = wcol + kb;
    v16h b;
#pragma unroll
    for (int j = 0; j < 16; ++j) b[j] = wp[j];

    acc = __builtin_amdgcn_wmma_f32_16x16x32_f16(false, a, false, b, (short)0, acc,
                                                 false, false);
  }
  // C layout: VGPR r -> row 8g + r, column = lane&15
#pragma unroll
  for (int r = 0; r < 8; ++r) {
    int orow = (tm << 4) + g * 8 + r;
    if (orow < M) Out[(size_t)orow * Ncols + col] = acc[r];
  }
}

// ---------- per-node attention coefficients ----------
__global__ void attn_coef_kernel(const float* __restrict__ h, const float* __restrict__ as,
                                 const float* __restrict__ ad, float* __restrict__ alpS,
                                 float* __restrict__ alpD, int Nn, int H) {
  int idx = blockIdx.x * blockDim.x + threadIdx.x;
  if (idx >= Nn * H) return;
  int n = idx / H, hh = idx - n * H;
  const float* hp  = h  + ((size_t)n * H + hh) * 32;
  const float* asp = as + hh * 32;
  const float* adp = ad + hh * 32;
  float s = 0.f, d = 0.f;
#pragma unroll
  for (int c = 0; c < 32; ++c) { s += hp[c] * asp[c]; d += hp[c] * adp[c]; }
  alpS[idx] = s; alpD[idx] = d;
}

// ---------- edge pass 1: logits + segment max ----------
__global__ void edge_logit_max_kernel(const int* __restrict__ ei, const float* __restrict__ alpS,
                                      const float* __restrict__ alpD, float* __restrict__ ebuf,
                                      unsigned* __restrict__ m, int E, int Nn, int H) {
  int idx = blockIdx.x * blockDim.x + threadIdx.x;
  if (idx >= (E + Nn) * H) return;
  int e = idx / H, h = idx - e * H;
  int src, dst;
  if (e < E) { src = ei[e]; dst = ei[E + e]; } else { src = dst = e - E; }
  float v = alpS[src * H + h] + alpD[dst * H + h];
  float l = (v > 0.f) ? v : NEG_SLOPE * v;
  ebuf[idx] = l;
  atomicMax(&m[dst * H + h], enc_f32(l));
}

// ---------- edge pass 2: exp(e - m[dst]) + denominator ----------
__global__ void edge_p_kernel(const int* __restrict__ ei, float* __restrict__ ebuf,
                              const unsigned* __restrict__ m, float* __restrict__ denom,
                              int E, int Nn, int H) {
  int idx = blockIdx.x * blockDim.x + threadIdx.x;
  if (idx >= (E + Nn) * H) return;
  int e = idx / H, h = idx - e * H;
  int dst = (e < E) ? ei[E + e] : (e - E);
  float mf = dec_f32(m[dst * H + h]);
  if (isinf(mf)) mf = 0.0f;
  float p = expf(ebuf[idx] - mf);
  ebuf[idx] = p;
  atomicAdd(&denom[dst * H + h], p);
}

// ---------- edge pass 3: weighted scatter of messages ----------
__global__ void edge_scatter_kernel(const int* __restrict__ ei, const float* __restrict__ ebuf,
                                    const float* __restrict__ denom, const float* __restrict__ hf,
                                    float* __restrict__ acc, int E, int Nn, int H) {
  int idx = blockIdx.x * blockDim.x + threadIdx.x;
  if (idx >= (E + Nn) * H) return;
  int e = idx / H, h = idx - e * H;
  int src, dst;
  if (e < E) { src = ei[e]; dst = ei[E + e]; } else { src = dst = e - E; }
  float alpha = ebuf[idx] / (denom[dst * H + h] + 1e-16f);
  const float4* hs = (const float4*)(hf + ((size_t)src * H + h) * 32);
  float* od = acc + ((size_t)dst * H + h) * 32;
#pragma unroll
  for (int q = 0; q < 8; ++q) {
    float4 v = hs[q];
    atomicAdd(od + 4 * q + 0, v.x * alpha);
    atomicAdd(od + 4 * q + 1, v.y * alpha);
    atomicAdd(od + 4 * q + 2, v.z * alpha);
    atomicAdd(od + 4 * q + 3, v.w * alpha);
  }
}

// ---------- bias + optional tanh ----------
__global__ void finalize_kernel(const float* __restrict__ acc, const float* __restrict__ bias,
                                float* __restrict__ out, int Nn, int HC, int doTanh) {
  int i = blockIdx.x * blockDim.x + threadIdx.x;
  if (i >= Nn * HC) return;
  int j = i % HC;
  float v = acc[i] + bias[j];
  if (doTanh) v = tanhf(v);
  out[i] = v;
}

// ---------- graph pooling ----------
__global__ void pool_kernel(const float* __restrict__ act, const int* __restrict__ batch,
                            unsigned* __restrict__ gmax, float* __restrict__ gsum,
                            float* __restrict__ cnt, int Nn) {
  int i = blockIdx.x * blockDim.x + threadIdx.x;
  if (i >= Nn * 32) return;
  int n = i >> 5, c = i & 31;
  int b = batch[n];
  float v = act[i];
  atomicMax(&gmax[b * 32 + c], enc_f32(v));
  atomicAdd(&gsum[b * 32 + c], v);
  if (c == 0) atomicAdd(&cnt[b], 1.0f);
}

// ---------- readout: feat=[max|mean], out = feat@Wo + bo ----------
__global__ void readout_kernel(const unsigned* __restrict__ gmax, const float* __restrict__ gsum,
                               const float* __restrict__ cnt, const float* __restrict__ Wo,
                               const float* __restrict__ bo, float* __restrict__ dout, int Bg) {
  int b = blockIdx.x * blockDim.x + threadIdx.x;
  if (b >= Bg) return;
  float feat[64];
#pragma unroll
  for (int c = 0; c < 32; ++c) feat[c] = dec_f32(gmax[b * 32 + c]);
  float invc = 1.0f / fmaxf(cnt[b], 1.0f);
#pragma unroll
  for (int c = 0; c < 32; ++c) feat[32 + c] = gsum[b * 32 + c] * invc;
  float o = bo[0];
#pragma unroll
  for (int j = 0; j < 64; ++j) o += feat[j] * Wo[j];
  dout[b] = o;                                  // out: [B,1]
  float* fo = dout + Bg + (size_t)b * 64;       // feat: [B,64]
#pragma unroll
  for (int j = 0; j < 64; ++j) fo[j] = feat[j];
}

// ============================ host launcher ============================
extern "C" void kernel_launch(void* const* d_in, const int* in_sizes, int n_in,
                              void* d_out, int out_size, void* d_ws, size_t ws_size,
                              hipStream_t stream) {
  (void)n_in; (void)out_size; (void)ws_size;
  const float* x   = (const float*)d_in[0];
  const int*   ei  = (const int*)d_in[1];
  const int*   bat = (const int*)d_in[2];
  const float* W1  = (const float*)d_in[3];
  const float* as1 = (const float*)d_in[4];
  const float* ad1 = (const float*)d_in[5];
  const float* b1  = (const float*)d_in[6];
  const float* W2  = (const float*)d_in[7];
  const float* as2 = (const float*)d_in[8];
  const float* ad2 = (const float*)d_in[9];
  const float* b2  = (const float*)d_in[10];
  const float* W3  = (const float*)d_in[11];
  const float* as3 = (const float*)d_in[12];
  const float* ad3 = (const float*)d_in[13];
  const float* b3  = (const float*)d_in[14];
  const float* Wo  = (const float*)d_in[15];
  const float* bo  = (const float*)d_in[16];

  const int N  = in_sizes[0] / 5;
  const int E  = in_sizes[1] / 2;
  const int Bg = 512;
  const int EP = E + N;

  // --- carve workspace ---
  char* wp = (char*)d_ws;
  auto carve = [&](size_t bytes) -> char* {
    char* r = wp; wp += (bytes + 255) & ~(size_t)255; return r;
  };
  float*    bufA  = (float*)carve((size_t)N * 256 * 4);  // activations (in/out)
  float*    bufH  = (float*)carve((size_t)N * 256 * 4);  // h = x @ W
  float*    bufC  = (float*)carve((size_t)N * 256 * 4);  // aggregation accumulator
  _Float16* x16   = (_Float16*)carve((size_t)N * 256 * 2);
  _Float16* wt16  = (_Float16*)carve((size_t)256 * 256 * 2);
  float*    alpS  = (float*)carve((size_t)N * 8 * 4);
  float*    alpD  = (float*)carve((size_t)N * 8 * 4);
  unsigned* mbuf  = (unsigned*)carve((size_t)N * 8 * 4);
  float*    denom = (float*)carve((size_t)N * 8 * 4);
  float*    ebuf  = (float*)carve((size_t)EP * 8 * 4);
  unsigned* gmax  = (unsigned*)carve((size_t)Bg * 32 * 4);
  float*    gsum  = (float*)carve((size_t)Bg * 32 * 4);
  float*    cnt   = (float*)carve((size_t)Bg * 4);

  const int TB = 256;
  auto cdiv = [](long long a, long long b) { return (int)((a + b - 1) / b); };

  auto run_layer = [&](const float* Xsrc, int Kin, int Kpad, int H,
                       const float* W, const float* as, const float* ad, const float* bias,
                       float* actOut, int doTanh) {
    const int HC = H * 32;
    cvt_wt_f16_kernel<<<cdiv((long long)HC * Kpad, TB), TB, 0, stream>>>(W, wt16, Kin, Kpad, HC);
    cvt_x_f16_kernel<<<cdiv((long long)N * Kpad, TB), TB, 0, stream>>>(Xsrc, x16, N, Kin, Kpad);
    const int tiles = ((N + 15) / 16) * (HC / 16);
    gemm_wmma_f16_kernel<<<cdiv(tiles, 8), TB, 0, stream>>>(x16, wt16, bufH, N, Kpad, HC);
    attn_coef_kernel<<<cdiv((long long)N * H, TB), TB, 0, stream>>>(bufH, as, ad, alpS, alpD, N, H);
    fill_u32_kernel<<<cdiv((long long)N * H, TB), TB, 0, stream>>>(mbuf, ENC_NEG_INF, N * H);
    hipMemsetAsync(denom, 0, (size_t)N * H * 4, stream);
    hipMemsetAsync(bufC, 0, (size_t)N * HC * 4, stream);
    edge_logit_max_kernel<<<cdiv((long long)EP * H, TB), TB, 0, stream>>>(ei, alpS, alpD, ebuf, mbuf, E, N, H);
    edge_p_kernel<<<cdiv((long long)EP * H, TB), TB, 0, stream>>>(ei, ebuf, mbuf, denom, E, N, H);
    edge_scatter_kernel<<<cdiv((long long)EP * H, TB), TB, 0, stream>>>(ei, ebuf, denom, bufH, bufC, E, N, H);
    finalize_kernel<<<cdiv((long long)N * HC, TB), TB, 0, stream>>>(bufC, bias, actOut, N, HC, doTanh);
  };

  // layer 1: 5 -> 8x32, tanh ; layer 2: 256 -> 2x32, tanh ; layer 3: 64 -> 1x32
  run_layer(x,    5,  32, 8, W1, as1, ad1, b1, bufA, 1);
  run_layer(bufA, 256, 256, 2, W2, as2, ad2, b2, bufA, 1);
  run_layer(bufA, 64,  64, 1, W3, as3, ad3, b3, bufA, 0);

  // pooling + readout
  fill_u32_kernel<<<cdiv(Bg * 32, TB), TB, 0, stream>>>(gmax, ENC_NEG_INF, Bg * 32);
  hipMemsetAsync(gsum, 0, (size_t)Bg * 32 * 4, stream);
  hipMemsetAsync(cnt, 0, (size_t)Bg * 4, stream);
  pool_kernel<<<cdiv((long long)N * 32, TB), TB, 0, stream>>>(bufA, bat, gmax, gsum, cnt, N);
  readout_kernel<<<cdiv(Bg, TB), TB, 0, stream>>>(gmax, gsum, cnt, Wo, bo, (float*)d_out, Bg);
}
